// LocalCorrelation_14542759264973
// MI455X (gfx1250) — matrix operand order repaired and checked by reading, not compiled, not run on hardware
//
#include <hip/hip_runtime.h>

// ---- types for CDNA5 WMMA ----
typedef _Float16 v16h __attribute__((ext_vector_type(16)));
typedef _Float16 v8h  __attribute__((ext_vector_type(8)));
typedef _Float16 v2h  __attribute__((ext_vector_type(2)));
typedef float    v8f  __attribute__((ext_vector_type(8)));

#define RADIUS 4
#define DD     9              // 2*RADIUS+1
#define ND     81             // DD*DD
#define CCH    256            // channels (K of the GEMM)
#define HH     96
#define WW     96
#define HW     (HH*WW)
#define TS     16             // s-tile side per workgroup
#define TR     24             // t-region side = TS + 2*RADIUS
#define KC     64             // channel chunk staged in LDS
#define SSTR   72             // LDS stride per pixel (halves): KC + 8 pad (bank spread)
#define NS_PX  (TS*TS)        // 256 s pixels
#define NT_PX  (TR*TR)        // 576 t pixels

__global__ __launch_bounds__(256)
void local_corr_wmma_f16(const float* __restrict__ fs,
                         const float* __restrict__ ft,
                         float* __restrict__ out) {
    // 119,808 B of LDS: s tile then t region, [pixel][channel] f16
    __shared__ _Float16 lds[(NS_PX + NT_PX) * SSTR];
    _Float16* lds_s = lds;
    _Float16* lds_t = lds + NS_PX * SSTR;

    const int tid  = threadIdx.x;
    const int lane = tid & 31;     // wave32
    const int wv   = tid >> 5;     // 8 waves
    const int lh   = lane >> 4;    // lane half (0/1) per WMMA striping
    const int m    = lane & 15;    // M/N selector within 16x16 tile

    const int bb = blockIdx.y;                 // batch
    const int tb = blockIdx.x;                 // 0..35 spatial tile
    const int h0 = (tb / (WW / TS)) * TS;
    const int w0 = (tb % (WW / TS)) * TS;

    // accumulators: 2 M-tiles/wave x 9 N-tiles, f32
    v8f acc[2][9];
    {
        v8f z = {};
#pragma unroll
        for (int t = 0; t < 2; ++t)
#pragma unroll
            for (int n = 0; n < 9; ++n) acc[t][n] = z;
    }

    // staging mapping: thread <-> one s pixel of the 16x16 tile
    const int spy = tid >> 4, spx = tid & 15;
    const float* gs = fs + (size_t)bb * CCH * HW + (size_t)(h0 + spy) * WW + (w0 + spx);
    const float* gt_base = ft + (size_t)bb * CCH * HW;

    for (int chunk = 0; chunk < CCH / KC; ++chunk) {
        const int c0 = chunk * KC;

        // ---- stage s: f32 global (coalesced over w) -> f16 LDS [pixel][channel] ----
#pragma unroll 4
        for (int cc = 0; cc < KC; cc += 2) {
            float x0 = gs[(size_t)(c0 + cc) * HW];
            float x1 = gs[(size_t)(c0 + cc + 1) * HW];
            v2h p; p.x = (_Float16)x0; p.y = (_Float16)x1;
            *(v2h*)&lds_s[tid * SSTR + cc] = p;
        }
        // ---- stage t region 24x24 with zero padding outside the image ----
        for (int p = tid; p < NT_PX; p += 256) {
            const int trow = p / TR, tcol = p % TR;
            const int gh = h0 - RADIUS + trow;
            const int gw = w0 - RADIUS + tcol;
            const bool inb = ((unsigned)gh < HH) && ((unsigned)gw < WW);
            const float* gt = gt_base + (size_t)gh * WW + gw;
#pragma unroll 4
            for (int cc = 0; cc < KC; cc += 2) {
                float x0 = inb ? gt[(size_t)(c0 + cc) * HW] : 0.0f;
                float x1 = inb ? gt[(size_t)(c0 + cc + 1) * HW] : 0.0f;
                v2h q; q.x = (_Float16)x0; q.y = (_Float16)x1;
                *(v2h*)&lds_t[p * SSTR + cc] = q;
            }
        }
        __syncthreads();

        // ---- compute: each wave owns 2 M-tiles (4x4 s patches) ----
#pragma unroll
        for (int t = 0; t < 2; ++t) {
            const int mt = 2 * wv + t;
            const int ti = mt >> 2, tj = mt & 3;       // patch coords in 4x4 grid
            const int px_s = (4 * ti + (m >> 2)) * TS + 4 * tj + (m & 3);
#pragma unroll
            for (int kl = 0; kl < KC; kl += 32) {
                // A fragment (16x32 f16): lane<16 holds K 0-7 & 16-23, lane>=16 K 8-15 & 24-31
                const int abase = px_s * SSTR + kl + 8 * lh;
                v8h alo = *(const v8h*)&lds_s[abase];
                v8h ahi = *(const v8h*)&lds_s[abase + 16];
                v16h af = __builtin_shufflevector(alo, ahi,
                    0,1,2,3,4,5,6,7,8,9,10,11,12,13,14,15);
#pragma unroll
                for (int di = 0; di < 3; ++di)
#pragma unroll
                    for (int dj = 0; dj < 3; ++dj) {
                        // B fragment (32x16 f16): lane<16 holds K 0-7 & 16-23, lane>=16 K 8-15 & 24-31
                        const int px_t = (4 * (ti + di) + (m >> 2)) * TR
                                       + 4 * (tj + dj) + (m & 3);
                        const int bbase = px_t * SSTR + kl + 8 * lh;
                        v8h blo = *(const v8h*)&lds_t[bbase];
                        v8h bhi = *(const v8h*)&lds_t[bbase + 16];
                        v16h bf = __builtin_shufflevector(blo, bhi,
                            0,1,2,3,4,5,6,7,8,9,10,11,12,13,14,15);
                        acc[t][di * 3 + dj] = __builtin_amdgcn_wmma_f32_16x16x32_f16(
                            false, af, false, bf, (short)0, acc[t][di * 3 + dj],
                            false, false);
                    }
            }
        }
        __syncthreads();
    }

    // ---- scatter valid band elements into LDS as [d][pixel] f32 ----
    float* lds_o = (float*)lds;   // 81*256 floats = 82,944 B, fits in the LDS buffer
#pragma unroll
    for (int t = 0; t < 2; ++t) {
        const int mt = 2 * wv + t;
        const int ti = mt >> 2, tj = mt & 3;
#pragma unroll
        for (int di = 0; di < 3; ++di)
#pragma unroll
            for (int dj = 0; dj < 3; ++dj)
#pragma unroll
                for (int r = 0; r < 8; ++r) {
                    // C/D layout: VGPR r, lanes 0-15 -> M=r, lanes 16-31 -> M=r+8; N = lane&15
                    const int M = r + 8 * lh;
                    const int N = m;
                    const int dy = 4 * di + (N >> 2) - (M >> 2);
                    const int dx = 4 * dj + (N & 3) - (M & 3);
                    if (((unsigned)dy < DD) && ((unsigned)dx < DD)) {
                        const int srow = 4 * ti + (M >> 2);
                        const int scol = 4 * tj + (M & 3);
                        lds_o[(dy * DD + dx) * NS_PX + srow * TS + scol] =
                            acc[t][di * 3 + dj][r] * (1.0f / CCH);
                    }
                }
    }
    __syncthreads();

    // ---- coalesced writeout: per displacement plane, 16x16 pixels ----
    float* po = out + (size_t)bb * ND * HW + (size_t)(h0 + spy) * WW + (w0 + spx);
#pragma unroll 3
    for (int d = 0; d < ND; ++d) {
        po[(size_t)d * HW] = lds_o[d * NS_PX + tid];
    }
}

extern "C" void kernel_launch(void* const* d_in, const int* in_sizes, int n_in,
                              void* d_out, int out_size, void* d_ws, size_t ws_size,
                              hipStream_t stream) {
    const float* fs = (const float*)d_in[0];
    const float* ft = (const float*)d_in[1];
    float* out = (float*)d_out;
    const int Bn = in_sizes[0] / (CCH * HW);          // 16
    dim3 grid((HH / TS) * (WW / TS), Bn);             // (36, 16)
    local_corr_wmma_f16<<<grid, dim3(256), 0, stream>>>(fs, ft, out);
}
